// GlobalInteractor_hivt_50697793962574
// MI455X (gfx1250) — matrix operand order repaired and checked by reading, not compiled, not run on hardware
//
#include <hip/hip_runtime.h>
#include <hip/hip_bf16.h>

// ---------------------------------------------------------------------------
// HiVT GlobalInteractor forward for MI455X (gfx1250, wave32, WMMA).
// All dense GEMMs run through v_wmma_f32_16x16x32_f16 with f16 operands
// staged in LDS (A row-major, B transposed for contiguous per-lane fragment
// reads) and f32 accumulation. Segment softmax uses global float atomics.
// ---------------------------------------------------------------------------

typedef __attribute__((ext_vector_type(16))) _Float16 v16h;
typedef __attribute__((ext_vector_type(8)))  _Float16 v8h;
typedef __attribute__((ext_vector_type(4)))  _Float16 v4h;
typedef __attribute__((ext_vector_type(8)))  float    v8f;

#define BM 32
#define BN 64
#define BK 32
#define LPAD 56   // LDS row stride in halves: 112B = 28 banks -> conflict-free

union H16 { v16h v16; v8h v8[2]; };

// ---------------------------------------------------------------------------
// GEMM: out[R,NCOLS] (+)= RELU?(X[R,K] @ W[K,NCOLS] + bias)
// 256 threads = 8 waves; wave w -> rowTile w/4 (0..1), colTile w%4 (0..3).
// Each wave computes one 16x16 f32 tile via V_WMMA_F32_16X16X32_F16.
// K, NCOLS, RELU, ACC compile-time -> fully unrolled, branch-free epilogue.
// NOTE: all launched R are multiples of BM (20000, 640000), grids exact.
// ---------------------------------------------------------------------------
template<int K, int NCOLS, int RELU, int ACC>
__global__ __launch_bounds__(256) void gemm_wmma_kernel(
    const float* __restrict__ X, const float* __restrict__ W,
    const float* __restrict__ bias, float* __restrict__ out, int R)
{
  __shared__ alignas(16) _Float16 As [BM][LPAD];   // A tile, row-major
  __shared__ alignas(16) _Float16 BsT[BN][LPAD];   // B tile, transposed [n][k]

  const int tid  = threadIdx.x;
  const int lane = tid & 31;
  const int wave = tid >> 5;
  const int rowTile = wave >> 2;   // 0..1
  const int colTile = wave & 3;    // 0..3
  const int row0 = blockIdx.x * BM;
  const int col0 = blockIdx.y * BN;

  // Fragment geometry (ISA 7.12.2 lane layouts)
  const int mRow = rowTile * 16 + (lane & 15);
  const int aSel = (lane >= 16) ? 8 : 0;     // A: halves 0-7 -> K=aSel.., 8-15 -> K=16+aSel..
  const int nCol = colTile * 16 + (lane & 15);
  const int bSel = (lane >= 16) ? 8 : 0;     // B: halves 0-7 -> K=bSel.., 8-15 -> K=16+bSel..

  v8f c = {};

  #pragma unroll
  for (int k0 = 0; k0 < K; k0 += BK) {
    // ---- stage A (32x32): each thread one float4 -> 4 halves (b128 + b64) ----
    {
      int r  = tid >> 3;            // 0..31
      int kq = (tid & 7) << 2;      // 0,4,..,28
      int gr = min(row0 + r, R - 1);          // branchless tail clamp
      const float4 xv = *(const float4*)&X[(size_t)gr * K + k0 + kq];
      v4h hv; hv[0] = (_Float16)xv.x; hv[1] = (_Float16)xv.y;
              hv[2] = (_Float16)xv.z; hv[3] = (_Float16)xv.w;
      *(v4h*)&As[r][kq] = hv;
    }
    // ---- stage B (32x64) transposed: float4 along N -> 4 scattered b16 ----
    #pragma unroll
    for (int cpass = 0; cpass < 2; ++cpass) {
      int idx = tid + cpass * 256;  // 0..511
      int k   = idx >> 4;           // 0..31
      int n4  = (idx & 15) << 2;    // 0,4,..,60
      const float4 wv = *(const float4*)&W[(size_t)(k0 + k) * NCOLS + col0 + n4];
      BsT[n4 + 0][k] = (_Float16)wv.x;
      BsT[n4 + 1][k] = (_Float16)wv.y;
      BsT[n4 + 2][k] = (_Float16)wv.z;
      BsT[n4 + 3][k] = (_Float16)wv.w;
    }
    if (k0 + BK < K) {
      // prefetch next W tile -> global_prefetch_b8
      __builtin_prefetch(&W[(size_t)(k0 + BK) * NCOLS + col0 + (tid & 63)], 0, 1);
    }
    __syncthreads();

    // ---- fragments: 4 contiguous 16B LDS reads per lane ----
    H16 ha, hb;
    ha.v8[0] = *(const v8h*)&As [mRow][aSel];
    ha.v8[1] = *(const v8h*)&As [mRow][16 + aSel];
    hb.v8[0] = *(const v8h*)&BsT[nCol][bSel];
    hb.v8[1] = *(const v8h*)&BsT[nCol][16 + bSel];

    c = __builtin_amdgcn_wmma_f32_16x16x32_f16(
        /*neg_a=*/false, ha.v16, /*neg_b=*/false, hb.v16,
        /*c_mod=*/(short)0, c, /*reuse_a=*/false, /*reuse_b=*/false);
    __syncthreads();
  }

  // ---- epilogue: C/D layout M = r + (lane>=16 ? 8 : 0), N = lane&15 ----
  const int mBase = row0 + rowTile * 16 + ((lane >= 16) ? 8 : 0);
  const int col   = col0 + nCol;
  const float bv  = bias ? bias[col] : 0.0f;
  #pragma unroll
  for (int r = 0; r < 8; ++r) {
    float v = c[r] + bv;
    if (RELU) v = fmaxf(v, 0.0f);
    float* p = &out[(size_t)(mBase + r) * NCOLS + col];
    if (ACC) *p += v; else *p = v;
  }
}

// ---------------------------------------------------------------------------
// Layernorm over rows of 64: one wave32 per row, 2 elems per lane.
// ---------------------------------------------------------------------------
__global__ __launch_bounds__(256) void ln64_kernel(
    const float* __restrict__ in, const float* __restrict__ g,
    const float* __restrict__ b, float* __restrict__ out, int R, int relu)
{
  int wave = threadIdx.x >> 5, lane = threadIdx.x & 31;
  int row = blockIdx.x * 8 + wave;
  if (row >= R) return;
  float x0 = in[row * 64 + lane];
  float x1 = in[row * 64 + 32 + lane];
  float s = x0 + x1;
  #pragma unroll
  for (int m = 16; m; m >>= 1) s += __shfl_xor(s, m, 32);
  float mu = s * (1.0f / 64.0f);
  float d0 = x0 - mu, d1 = x1 - mu;
  float v = d0 * d0 + d1 * d1;
  #pragma unroll
  for (int m = 16; m; m >>= 1) v += __shfl_xor(v, m, 32);
  float inv = rsqrtf(v * (1.0f / 64.0f) + 1e-5f);
  float y0 = d0 * inv * g[lane] + b[lane];
  float y1 = d1 * inv * g[lane + 32] + b[lane + 32];
  if (relu) { y0 = fmaxf(y0, 0.0f); y1 = fmaxf(y1, 0.0f); }
  out[row * 64 + lane]      = y0;
  out[row * 64 + 32 + lane] = y1;
}

// ---------------------------------------------------------------------------
// Edge front: rel-pose geometry + the two tiny 2->64 linears + LN + relu.
// One wave32 per edge. Outputs u0,u1 [E,64] (inputs to the W2 GEMMs).
// ---------------------------------------------------------------------------
__device__ inline void embed_front_branch(
    float x0, float x1, const float* __restrict__ W1, const float* __restrict__ b1,
    const float* __restrict__ g, const float* __restrict__ be,
    int lane, float* __restrict__ out, int e)
{
  int dA = lane, dB = lane + 32;
  float hA = x0 * W1[dA] + x1 * W1[64 + dA] + b1[dA];
  float hB = x0 * W1[dB] + x1 * W1[64 + dB] + b1[dB];
  float s = hA + hB;
  #pragma unroll
  for (int m = 16; m; m >>= 1) s += __shfl_xor(s, m, 32);
  float mu = s * (1.0f / 64.0f);
  float a = hA - mu, bb = hB - mu;
  float v = a * a + bb * bb;
  #pragma unroll
  for (int m = 16; m; m >>= 1) v += __shfl_xor(v, m, 32);
  float inv = rsqrtf(v * (1.0f / 64.0f) + 1e-5f);
  out[e * 64 + dA] = fmaxf(a * inv * g[dA] + be[dA], 0.0f);
  out[e * 64 + dB] = fmaxf(bb * inv * g[dB] + be[dB], 0.0f);
}

__global__ __launch_bounds__(256) void edge_front_kernel(
    const float* __restrict__ positions, const float* __restrict__ rmat,
    const float* __restrict__ ang, const int* __restrict__ ei,
    const float* W1a, const float* b1a, const float* g1a, const float* be1a,
    const float* W1b, const float* b1b, const float* g1b, const float* be1b,
    float* __restrict__ u0, float* __restrict__ u1, int E)
{
  int wave = threadIdx.x >> 5, lane = threadIdx.x & 31;
  int e = blockIdx.x * 8 + wave;
  if (e >= E) return;
  int src = ei[e], dst = ei[E + e];
  // positions[:, T-1] with T=20 -> offsets 38,39
  float px = positions[src * 40 + 38] - positions[dst * 40 + 38];
  float py = positions[src * 40 + 39] - positions[dst * 40 + 39];
  // out[j] = sum_i rel[i] * R[dst][i][j]
  float rx = px * rmat[dst * 4 + 0] + py * rmat[dst * 4 + 2];
  float ry = px * rmat[dst * 4 + 1] + py * rmat[dst * 4 + 3];
  float th = ang[src] - ang[dst];
  float tc = cosf(th), ts = sinf(th);
  embed_front_branch(rx, ry, W1a, b1a, g1a, be1a, lane, u0, e);
  embed_front_branch(tc, ts, W1b, b1b, g1b, be1b, lane, u1, e);
}

// ---------------------------------------------------------------------------
// Attention: segment softmax over destinations via global float atomics.
// ---------------------------------------------------------------------------
__device__ inline void atomicMaxFloat(float* addr, float val)
{
  if (val >= 0.0f) atomicMax((int*)addr, __float_as_int(val));
  else             atomicMin((unsigned int*)addr, __float_as_uint(val));
}

__global__ __launch_bounds__(256) void attn_init_kernel(
    float* __restrict__ mmax, float* __restrict__ denom,
    float* __restrict__ agg, int N)
{
  int idx = blockIdx.x * 256 + threadIdx.x;
  if (idx < N * 64) agg[idx] = 0.0f;
  if (idx < N * 8) { mmax[idx] = -__builtin_inff(); denom[idx] = 0.0f; }
}

__global__ __launch_bounds__(256) void attn_logits_kernel(
    const float* __restrict__ qn, const float* __restrict__ kn,
    const float* __restrict__ ke, const int* __restrict__ ei,
    float* __restrict__ alpha, float* __restrict__ mmax, int E)
{
  int idx = blockIdx.x * 256 + threadIdx.x;
  if (idx >= E * 8) return;
  int e = idx >> 3, h = idx & 7;
  int src = ei[e], dst = ei[E + e];
  const float* q  = qn + dst * 64 + h * 8;
  const float* k  = kn + src * 64 + h * 8;
  const float* kp = ke + (size_t)e * 64 + h * 8;
  float s = 0.0f;
  #pragma unroll
  for (int i = 0; i < 8; ++i) s += q[i] * (k[i] + kp[i]);
  s *= 0.35355339059327373f;   // 1/sqrt(Dh), Dh=8
  alpha[idx] = s;
  atomicMaxFloat(&mmax[dst * 8 + h], s);
}

__global__ __launch_bounds__(256) void attn_exp_kernel(
    float* __restrict__ alpha, const float* __restrict__ mmax,
    float* __restrict__ denom, const int* __restrict__ ei, int E)
{
  int idx = blockIdx.x * 256 + threadIdx.x;
  if (idx >= E * 8) return;
  int e = idx >> 3, h = idx & 7;
  int dst = ei[E + e];
  float m = mmax[dst * 8 + h];
  if (!(fabsf(m) < 1e38f)) m = 0.0f;        // where(isfinite(mmax), mmax, 0)
  float ex = __expf(alpha[idx] - m);
  alpha[idx] = ex;
  atomicAdd(&denom[dst * 8 + h], ex);
}

__global__ __launch_bounds__(256) void attn_agg_kernel(
    const float* __restrict__ alpha, const float* __restrict__ denom,
    const float* __restrict__ vn, const float* __restrict__ ve,
    const int* __restrict__ ei, float* __restrict__ agg, int E)
{
  int idx = blockIdx.x * 256 + threadIdx.x;
  if (idx >= E * 64) return;
  int e = idx >> 6, d = idx & 63, h = d >> 3;
  int src = ei[e], dst = ei[E + e];
  float w = alpha[e * 8 + h] / (denom[dst * 8 + h] + 1e-16f);
  atomicAdd(&agg[dst * 64 + d], w * (vn[src * 64 + d] + ve[(size_t)e * 64 + d]));
}

// GRU-style gated update: t2 = agg + sigmoid(pre) * (selfv - agg)
__global__ __launch_bounds__(256) void gate_kernel(
    const float* __restrict__ pre, const float* __restrict__ selfv,
    const float* __restrict__ agg, float* __restrict__ t2, int total)
{
  int idx = blockIdx.x * 256 + threadIdx.x;
  if (idx >= total) return;
  float g = 1.0f / (1.0f + __expf(-pre[idx]));
  float a = agg[idx];
  t2[idx] = a + g * (selfv[idx] - a);
}

__global__ __launch_bounds__(256) void copy_kernel(
    const float* __restrict__ in, float* __restrict__ out, int total)
{
  int idx = blockIdx.x * 256 + threadIdx.x;
  if (idx < total) out[idx] = in[idx];
}

// out[m*N*64 + n*64 + d] = in[n*384 + m*64 + d]   (N,M,D) -> (M,N,D)
__global__ __launch_bounds__(256) void permute_kernel(
    const float* __restrict__ in, float* __restrict__ out, int N)
{
  int idx = blockIdx.x * 256 + threadIdx.x;
  if (idx >= N * 384) return;
  int n = idx / 384, c = idx % 384;
  int m = c >> 6, d = c & 63;
  out[(m * N + n) * 64 + d] = in[idx];
}

// ---------------------------------------------------------------------------
// Host-side orchestration
// ---------------------------------------------------------------------------
struct LayerP {
  const float *Whh,*Wih,*Wke,*Wkn,*Wm1,*Wm2,*Wout,*Wq,*Wself,*Wve,*Wvn;
  const float *bhh,*bih,*bke,*bkn,*bm1,*bm2,*bout,*bq,*bself,*bve,*bvn;
  const float *n1b,*n1g,*n2b,*n2g;
};

template<int K, int NCOLS, int RELU, int ACC>
static inline void gemm(const float* X, const float* W, const float* b, float* out,
                        int R, hipStream_t s)
{
  dim3 grid((R + BM - 1) / BM, NCOLS / BN);
  gemm_wmma_kernel<K, NCOLS, RELU, ACC><<<grid, 256, 0, s>>>(X, W, b, out, R);
}

extern "C" void kernel_launch(void* const* d_in, const int* in_sizes, int n_in,
                              void* d_out, int out_size, void* d_ws, size_t ws_size,
                              hipStream_t stream)
{
  const float* local_embed = (const float*)d_in[0];
  const float* positions   = (const float*)d_in[1];
  const float* rotate_mat  = (const float*)d_in[2];
  const float* rotate_ang  = (const float*)d_in[3];
  const int*   edge_index  = (const int*)d_in[4];

  const int N = in_sizes[0] / 64;
  const int E = in_sizes[4] / 2;

  // params flattened as jax pytree leaves: dict keys sorted lexicographically.
  int p = 5;
  auto F = [&](void) { return (const float*)d_in[p++]; };
  // aggr: W, b, b2, bw, g, g2
  const float *aggr_W = F(), *aggr_b = F(), *aggr_b2 = F(), *aggr_bw = F(),
              *aggr_g = F(), *aggr_g2 = F();
  // emb0: W1, W2, b1, b2, be1, g1
  const float *e0_W1 = F(), *e0_W2 = F(), *e0_b1 = F(), *e0_b2 = F(),
              *e0_be1 = F(), *e0_g1 = F();
  // emb1
  const float *e1_W1 = F(), *e1_W2 = F(), *e1_b1 = F(), *e1_b2 = F(),
              *e1_be1 = F(), *e1_g1 = F();
  // final: W, b, bw, g
  const float *fin_W = F(), *fin_b = F(), *fin_bw = F(), *fin_g = F();
  // layers: sorted -> W-group, b-group, n-group
  LayerP L[3];
  for (int l = 0; l < 3; ++l) {
    L[l].Whh = F(); L[l].Wih = F(); L[l].Wke = F(); L[l].Wkn = F();
    L[l].Wm1 = F(); L[l].Wm2 = F(); L[l].Wout = F(); L[l].Wq = F();
    L[l].Wself = F(); L[l].Wve = F(); L[l].Wvn = F();
    L[l].bhh = F(); L[l].bih = F(); L[l].bke = F(); L[l].bkn = F();
    L[l].bm1 = F(); L[l].bm2 = F(); L[l].bout = F(); L[l].bq = F();
    L[l].bself = F(); L[l].bve = F(); L[l].bvn = F();
    L[l].n1b = F(); L[l].n1g = F(); L[l].n2b = F(); L[l].n2g = F();
  }

  // ---- workspace layout (floats) ----
  size_t off = 0;
  auto A = [&](size_t nf) { float* q = (float*)d_ws + off; off += nf; return q; };
  const size_t E64 = (size_t)E * 64, N64 = (size_t)N * 64;
  float* eb0   = A(E64);          // u0 / ke
  float* eb1   = A(E64);          // u1 / ve
  float* ea    = A(E64);          // edge_attr (also temp h)
  float* alpha = A((size_t)E * 8);
  float* xbuf  = A(N64);
  float* xn    = A(N64);
  float* qn    = A(N64);
  float* kn    = A(N64);
  float* vn    = A(N64);
  float* t0    = A(N64);
  float* t1    = A(N64);
  float* t2    = A(N64);
  float* agg   = A(N64);
  float* ffb   = A((size_t)N * 256);
  float* mmax  = A((size_t)N * 8);
  float* denom = A((size_t)N * 8);
  float* outb  = A((size_t)N * 384);
  (void)ws_size; (void)n_in; (void)out_size;

  // ================= edge_attr pipeline =================
  edge_front_kernel<<<(E + 7) / 8, 256, 0, stream>>>(
      positions, rotate_mat, rotate_ang, edge_index,
      e0_W1, e0_b1, e0_g1, e0_be1, e1_W1, e1_b1, e1_g1, e1_be1,
      eb0, eb1, E);
  gemm<64, 64, 0, 0>(eb0, e0_W2, e0_b2, ea, E, stream);   // h  = u0@W2a + b2a
  gemm<64, 64, 0, 1>(eb1, e1_W2, e1_b2, ea, E, stream);   // h += u1@W2b + b2b
  ln64_kernel<<<(E + 7) / 8, 256, 0, stream>>>(ea, aggr_g, aggr_b, eb0, E, 1);
  gemm<64, 64, 0, 0>(eb0, aggr_W, aggr_bw, eb1, E, stream);
  ln64_kernel<<<(E + 7) / 8, 256, 0, stream>>>(eb1, aggr_g2, aggr_b2, ea, E, 0);

  // ================= x init =================
  copy_kernel<<<((int)N64 + 255) / 256, 256, 0, stream>>>(local_embed, xbuf, (int)N64);

  // ================= transformer layers =================
  for (int l = 0; l < 3; ++l) {
    const LayerP& P = L[l];
    ln64_kernel<<<(N + 7) / 8, 256, 0, stream>>>(xbuf, P.n1g, P.n1b, xn, N, 0);
    // node-level projections (q/k/v gathered per edge later)
    gemm<64, 64, 0, 0>(xn, P.Wq,  P.bq,  qn, N, stream);
    gemm<64, 64, 0, 0>(xn, P.Wkn, P.bkn, kn, N, stream);
    gemm<64, 64, 0, 0>(xn, P.Wvn, P.bvn, vn, N, stream);
    // edge-level projections
    gemm<64, 64, 0, 0>(ea, P.Wke, P.bke, eb0, E, stream);   // ke
    gemm<64, 64, 0, 0>(ea, P.Wve, P.bve, eb1, E, stream);   // ve
    // segment softmax attention
    attn_init_kernel<<<((int)N64 + 255) / 256, 256, 0, stream>>>(mmax, denom, agg, N);
    attn_logits_kernel<<<(E * 8 + 255) / 256, 256, 0, stream>>>(qn, kn, eb0, edge_index, alpha, mmax, E);
    attn_exp_kernel<<<(E * 8 + 255) / 256, 256, 0, stream>>>(alpha, mmax, denom, edge_index, E);
    attn_agg_kernel<<<((int)E64 + 255) / 256, 256, 0, stream>>>(alpha, denom, vn, eb1, edge_index, agg, E);
    // gated update
    gemm<64, 64, 0, 0>(agg, P.Wih, P.bih, t0, N, stream);
    gemm<64, 64, 0, 1>(xn,  P.Whh, P.bhh, t0, N, stream);
    gemm<64, 64, 0, 0>(xn,  P.Wself, P.bself, t1, N, stream);
    gate_kernel<<<((int)N64 + 255) / 256, 256, 0, stream>>>(t0, t1, agg, t2, (int)N64);
    gemm<64, 64, 0, 1>(t2, P.Wout, P.bout, xbuf, N, stream);   // x += upd@Wout+bout
    // feed-forward
    ln64_kernel<<<(N + 7) / 8, 256, 0, stream>>>(xbuf, P.n2g, P.n2b, t0, N, 0);
    gemm<64, 256, 1, 0>(t0, P.Wm1, P.bm1, ffb, N, stream);
    gemm<256, 64, 0, 1>(ffb, P.Wm2, P.bm2, xbuf, N, stream);
  }

  // ================= final head =================
  ln64_kernel<<<(N + 7) / 8, 256, 0, stream>>>(xbuf, fin_g, fin_b, t0, N, 0);
  gemm<64, 384, 0, 0>(t0, fin_W, fin_bw, outb, N, stream);
  permute_kernel<<<(N * 384 + 255) / 256, 256, 0, stream>>>(outb, (float*)d_out, N);
}
